// HGCN_19859928777301
// MI455X (gfx1250) — compile-verified, matrix operand order checked by reading, not run on hardware
//
#include <hip/hip_runtime.h>
#include <hip/hip_bf16.h>
#include <cstddef>

// ---------------------------------------------------------------------------
// Problem constants (match reference)
// ---------------------------------------------------------------------------
#define HDIM    128
#define NPAPER  100000
#define NAUTHOR 50000
#define NEDGE   600000
#define NLAYER  2

#define FLAG_RELU 1

typedef __attribute__((ext_vector_type(16))) __bf16 v16bf;
typedef __attribute__((ext_vector_type(8)))  float  v8f;

__device__ __forceinline__ v8f wmma_bf16(v16bf a, v16bf b, v8f c) {
  // D = A(16x32 bf16) * B(32x16 bf16) + C(16x16 f32), wave32
  return __builtin_amdgcn_wmma_f32_16x16x32_bf16(
      /*neg_a=*/false, a, /*neg_b=*/false, b,
      /*c_mod=*/(short)0, c, /*reuse_a=*/false, /*reuse_b=*/false);
}

// ---------------------------------------------------------------------------
// Weight pre-swizzle: f32 W[K,N] (optionally + W2) -> bf16 fragments laid out
// exactly as the 16-bit B-operand VGPR layout, so the GEMM can fetch a whole
// fragment with two ds_load_b128 per lane.
// ---------------------------------------------------------------------------
__global__ __launch_bounds__(256) void swizzle_w_kernel(
    const float* __restrict__ W, const float* __restrict__ W2,
    __bf16* __restrict__ dst, int K, int N)
{
  const int total = K * N;
  for (int i = blockIdx.x * blockDim.x + threadIdx.x; i < total;
       i += gridDim.x * blockDim.x) {
    const int k = i / N, n = i - k * N;
    float w = W[i];
    if (W2) w += W2[i];
    const int kb = k >> 5, kk = k & 31;
    const int half = (kk >> 3) & 1;
    const int j = (kk & 7) + ((kk >> 4) << 3);
    const int lane = half * 16 + (n & 15);
    const int nt = n >> 4;
    const int ntl = N >> 4;
    dst[(((size_t)(kb * ntl + nt) * 32 + lane) << 4) + j] = (__bf16)w;
  }
}

// Z[i] = X[i] + Y[i]  (combined bias for the fused paper update)
__global__ __launch_bounds__(256) void addvec_kernel(
    const float* __restrict__ X, const float* __restrict__ Y,
    float* __restrict__ Z, int n)
{
  int i = blockIdx.x * blockDim.x + threadIdx.x;
  if (i < n) Z[i] = X[i] + Y[i];
}

// ---------------------------------------------------------------------------
// Fused segmented WMMA GEMM.
//   C[M, NT*16] = act( sum_s (A_s[M,K_s] / rowcnt_s) @ Wfrag_seg_s + bias )
// 256 threads = 8 waves; each wave owns a 32-row x N output stripe:
// 2 A fragments x NT B tiles = 2*NT accumulators (16 for NT=8 -> 128 VGPRs).
// Per 32-deep k-step: 16 WMMAs vs 8 B-fragment LDS reads (each B reused 2x),
// with a one-deep register double-buffer on B.
// All WMMAs execute unconditionally with full EXEC (ISA requirement); the
// rare partial hi half-tile (M % 32 == 16, only the very last wave for
// M=50000) re-reads the lo rows and simply skips its stores.
// W is staged through LDS in 128-K chunks (NT=8 -> 32 KB of the 320 KB WGP
// LDS), coalesced uint4 global loads + ds stores.
// ---------------------------------------------------------------------------
template <int NT>
__global__ __launch_bounds__(256) void wmma_gemm_kernel(
    const float* __restrict__ A0, const float* __restrict__ rc0, int K0,
    const float* __restrict__ A1, const float* __restrict__ rc1, int K1,
    const float* __restrict__ A2, const float* __restrict__ rc2, int K2,
    const __bf16* __restrict__ Wfrag,
    const float* __restrict__ bias,   // nullable
    float* __restrict__ C, int M, int flags)
{
  __shared__ __align__(32) __bf16 lds[128 * NT * 16];   // one 128-K chunk of W

  const int tid   = threadIdx.x;
  const int lane  = tid & 31;
  const int wave  = tid >> 5;
  const int tileM = blockIdx.x * 8 + wave;  // 32-row tiles
  const int row0  = tileM * 32;
  const bool actLo = row0 < M;              // wave-uniform guards
  const bool actHi = (row0 + 16) < M;       // (M is a multiple of 16)
  const int nl   = lane & 15;
  const int half = lane >> 4;
  const int arow0 = row0 + nl;
  // Branch-free hi half: clamp to lo rows when the hi half-tile is past M so
  // every load stays in-bounds and every WMMA runs with EXEC all-1s; the
  // garbage results are dropped at store time.
  const int arow1 = actHi ? arow0 + 16 : arow0;
  const int N    = NT * 16;

  v8f acc0[NT], acc1[NT];
#pragma unroll
  for (int nt = 0; nt < NT; ++nt) {
    const float bv = bias ? bias[nt * 16 + nl] : 0.0f;
#pragma unroll
    for (int r = 0; r < 8; ++r) { acc0[nt][r] = bv; acc1[nt][r] = bv; }
  }

  const int Ktot = K0 + K1 + K2;
  for (int kc = 0; kc < Ktot; kc += 128) {
    // ---- stage this 128-K chunk of swizzled W into LDS (coalesced) ----
    {
      const uint4* src = (const uint4*)(Wfrag + ((size_t)(kc >> 5) * NT << 9));
      uint4* dst = (uint4*)lds;
      const int nvec = NT * 256;                    // 128*N bf16 / 8 per uint4
#pragma unroll 2
      for (int i = tid; i < nvec; i += 256) dst[i] = src[i];
    }
    __syncthreads();

    if (actLo) {
      // ---- select A segment for this chunk (segments are 128-multiples) ----
      const float* A; const float* rc; int segStart, segLen;
      if (kc < K0)            { A = A0; rc = rc0; segStart = 0;       segLen = K0; }
      else if (kc < K0 + K1)  { A = A1; rc = rc1; segStart = K0;      segLen = K1; }
      else                    { A = A2; rc = rc2; segStart = K0 + K1; segLen = K2; }

      float inv0 = 1.0f, inv1 = 1.0f;
      if (rc) {
        inv0 = 1.0f / fmaxf(rc[arow0], 1.0f);
        inv1 = 1.0f / fmaxf(rc[arow1], 1.0f);
      }
      const float4* Ar0 =
          (const float4*)(A + (size_t)arow0 * segLen + (kc - segStart));
      const float4* Ar1 =
          (const float4*)(A + (size_t)arow1 * segLen + (kc - segStart));

#pragma unroll
      for (int kb = 0; kb < 128; kb += 32) {
        const int k0 = kb + half * 8;
        v16bf av0, av1;
        {
          const float4 a0 = Ar0[(k0 >> 2) + 0];
          const float4 a1 = Ar0[(k0 >> 2) + 1];
          const float4 a2 = Ar0[(k0 >> 2) + 4];
          const float4 a3 = Ar0[(k0 >> 2) + 5];
          av0[0]  = (__bf16)(a0.x * inv0); av0[1]  = (__bf16)(a0.y * inv0);
          av0[2]  = (__bf16)(a0.z * inv0); av0[3]  = (__bf16)(a0.w * inv0);
          av0[4]  = (__bf16)(a1.x * inv0); av0[5]  = (__bf16)(a1.y * inv0);
          av0[6]  = (__bf16)(a1.z * inv0); av0[7]  = (__bf16)(a1.w * inv0);
          av0[8]  = (__bf16)(a2.x * inv0); av0[9]  = (__bf16)(a2.y * inv0);
          av0[10] = (__bf16)(a2.z * inv0); av0[11] = (__bf16)(a2.w * inv0);
          av0[12] = (__bf16)(a3.x * inv0); av0[13] = (__bf16)(a3.y * inv0);
          av0[14] = (__bf16)(a3.z * inv0); av0[15] = (__bf16)(a3.w * inv0);
        }
        {
          const float4 a0 = Ar1[(k0 >> 2) + 0];
          const float4 a1 = Ar1[(k0 >> 2) + 1];
          const float4 a2 = Ar1[(k0 >> 2) + 4];
          const float4 a3 = Ar1[(k0 >> 2) + 5];
          av1[0]  = (__bf16)(a0.x * inv1); av1[1]  = (__bf16)(a0.y * inv1);
          av1[2]  = (__bf16)(a0.z * inv1); av1[3]  = (__bf16)(a0.w * inv1);
          av1[4]  = (__bf16)(a1.x * inv1); av1[5]  = (__bf16)(a1.y * inv1);
          av1[6]  = (__bf16)(a1.z * inv1); av1[7]  = (__bf16)(a1.w * inv1);
          av1[8]  = (__bf16)(a2.x * inv1); av1[9]  = (__bf16)(a2.y * inv1);
          av1[10] = (__bf16)(a2.z * inv1); av1[11] = (__bf16)(a2.w * inv1);
          av1[12] = (__bf16)(a3.x * inv1); av1[13] = (__bf16)(a3.y * inv1);
          av1[14] = (__bf16)(a3.z * inv1); av1[15] = (__bf16)(a3.w * inv1);
        }

        // B fragments: register double-buffer over the NT tiles
        const __bf16* lbase = lds + (((kb >> 5) * NT * 32 + lane) << 4);
        v16bf bcur = *(const v16bf*)(lbase);
#pragma unroll
        for (int nt = 0; nt < NT; ++nt) {
          v16bf bnext;
          if (nt + 1 < NT)
            bnext = *(const v16bf*)(lbase + ((nt + 1) << 9));  // (nt+1)*32*16
          acc0[nt] = wmma_bf16(av0, bcur, acc0[nt]);
          acc1[nt] = wmma_bf16(av1, bcur, acc1[nt]);
          if (nt + 1 < NT) bcur = bnext;
        }
      }
    }
    __syncthreads();
  }

  if (actLo) {
#pragma unroll
    for (int nt = 0; nt < NT; ++nt) {
#pragma unroll
      for (int r = 0; r < 8; ++r) {
        float v = acc0[nt][r];
        if (flags & FLAG_RELU) v = fmaxf(v, 0.0f);
        C[(size_t)(row0 + half * 8 + r) * N + nt * 16 + nl] = v;
      }
    }
    if (actHi) {
#pragma unroll
      for (int nt = 0; nt < NT; ++nt) {
#pragma unroll
        for (int r = 0; r < 8; ++r) {
          float v = acc1[nt][r];
          if (flags & FLAG_RELU) v = fmaxf(v, 0.0f);
          C[(size_t)(row0 + 16 + half * 8 + r) * N + nt * 16 + nl] = v;
        }
      }
    }
  }
}

// ---------------------------------------------------------------------------
// Edge scatter-add: one wave per edge (grid-stride), lane owns a float4 slice.
// Sum tables (<=51MB) are L2-resident on the 192MB L2 -> on-die f32 atomics.
// ---------------------------------------------------------------------------
__global__ __launch_bounds__(256) void scatter_add_kernel(
    const float* __restrict__ feat,
    const int* __restrict__ src, const int* __restrict__ dst,
    float* __restrict__ sum, float* __restrict__ cnt, int E)
{
  const int lane = threadIdx.x & 31;
  const int wid  = (int)((blockIdx.x * blockDim.x + threadIdx.x) >> 5);
  const int nw   = (int)((gridDim.x * blockDim.x) >> 5);
  for (int e = wid; e < E; e += nw) {
    const int s = src[e];
    const int d = dst[e];
    float4 v = ((const float4*)(feat + (size_t)s * HDIM))[lane];
    float* out = sum + (size_t)d * HDIM + lane * 4;
    atomicAdd(out + 0, v.x);
    atomicAdd(out + 1, v.y);
    atomicAdd(out + 2, v.z);
    atomicAdd(out + 3, v.w);
    if (lane == 0) atomicAdd(cnt + d, 1.0f);
  }
}

// ---------------------------------------------------------------------------
static inline void launch_gemm8(const float* A0, const float* rc0, int K0,
                                const float* A1, const float* rc1, int K1,
                                const float* A2, const float* rc2, int K2,
                                const __bf16* Wf, const float* bias, float* C,
                                int M, int flags, hipStream_t s)
{
  dim3 grid(((M + 31) / 32 + 7) / 8);
  wmma_gemm_kernel<8><<<grid, 256, 0, s>>>(A0, rc0, K0, A1, rc1, K1,
                                           A2, rc2, K2, Wf, bias, C, M, flags);
}

extern "C" void kernel_launch(void* const* d_in, const int* in_sizes, int n_in,
                              void* d_out, int out_size, void* d_ws, size_t ws_size,
                              hipStream_t stream)
{
  const float* x_paper = (const float*)d_in[0];   // [NP,256]
  const float* x_author= (const float*)d_in[1];   // [NA,128]
  const float* lin_p_w = (const float*)d_in[2];   // [256,128]
  const float* lin_p_b = (const float*)d_in[3];   // [128]
  const float* lin_a_w = (const float*)d_in[4];   // [128,128]
  const float* lin_a_b = (const float*)d_in[5];   // [128]
  const float* Wl      = (const float*)d_in[6];   // [2,3,128,128]
  const float* bl      = (const float*)d_in[7];   // [2,3,128]
  const float* Wr      = (const float*)d_in[8];   // [2,3,128,128]
  const float* lin_w   = (const float*)d_in[9];   // [128,64]
  const float* lin_b   = (const float*)d_in[10];  // [64]
  const int* a2p_src   = (const int*)d_in[11];
  const int* a2p_dst   = (const int*)d_in[12];
  const int* p2a_src   = (const int*)d_in[13];
  const int* p2a_dst   = (const int*)d_in[14];
  const int* p2p_src   = (const int*)d_in[15];
  const int* p2p_dst   = (const int*)d_in[16];

  // ---- carve workspace (256B aligned chunks) ----
  char* p = (char*)d_ws;
  auto carve = [&](size_t bytes) -> void* {
    void* r = (void*)p;
    p += (bytes + 255) & ~(size_t)255;
    return r;
  };
  const size_t NPH = (size_t)NPAPER * HDIM * sizeof(float);
  const size_t NAH = (size_t)NAUTHOR * HDIM * sizeof(float);
  float* yp0 = (float*)carve(NPH);
  float* yp1 = (float*)carve(NPH);
  float* ya0 = (float*)carve(NAH);
  float* ya1 = (float*)carve(NAH);
  float* s1  = (float*)carve(NPH);                    // a2p sums -> paper
  float* s2  = (float*)carve(NPH);                    // p2p sums -> paper
  float* s3  = (float*)carve(NAH);                    // p2a sums -> author
  float* c1  = (float*)carve((size_t)NPAPER * sizeof(float));
  float* c2  = (float*)carve((size_t)NPAPER * sizeof(float));
  float* c3  = (float*)carve((size_t)NAUTHOR * sizeof(float));
  // swizzled bf16 weight fragments
  const size_t FR128 = (size_t)128 * 128;             // bf16 elems per 128x128
  __bf16* f_projp = (__bf16*)carve(2 * FR128 * 2);    // K=256,N=128
  __bf16* f_proja = (__bf16*)carve(FR128 * 2);        // K=128,N=128
  __bf16* f_pap[NLAYER]; __bf16* f_aut[NLAYER];
  for (int l = 0; l < NLAYER; ++l) {
    f_pap[l] = (__bf16*)carve(3 * FR128 * 2);         // [Wl0;Wl2;Wr0+Wr2]
    f_aut[l] = (__bf16*)carve(2 * FR128 * 2);         // [Wl1;Wr1]
  }
  __bf16* f_fin = (__bf16*)carve((size_t)128 * 64 * 2);   // K=128,N=64
  float* bsum[NLAYER];
  for (int l = 0; l < NLAYER; ++l) bsum[l] = (float*)carve(HDIM * sizeof(float));

  const dim3 swgrd(64), swblk(256);
  const dim3 sgrd(2048), sblk(256);

  // ---- one-time (per call) weight swizzles: all L2-resident, tiny ----
  swizzle_w_kernel<<<swgrd, swblk, 0, stream>>>(lin_p_w, nullptr, f_projp, 256, 128);
  swizzle_w_kernel<<<swgrd, swblk, 0, stream>>>(lin_a_w, nullptr, f_proja, 128, 128);
  for (int l = 0; l < NLAYER; ++l) {
    const float* Wl0 = Wl + (size_t)(l * 3 + 0) * FR128;
    const float* Wl1 = Wl + (size_t)(l * 3 + 1) * FR128;
    const float* Wl2 = Wl + (size_t)(l * 3 + 2) * FR128;
    const float* Wr0 = Wr + (size_t)(l * 3 + 0) * FR128;
    const float* Wr1 = Wr + (size_t)(l * 3 + 1) * FR128;
    const float* Wr2 = Wr + (size_t)(l * 3 + 2) * FR128;
    const size_t SEG = (size_t)4 * 8 * 512;           // frag elems per 128x128
    swizzle_w_kernel<<<swgrd, swblk, 0, stream>>>(Wl0, nullptr, f_pap[l] + 0 * SEG, 128, 128);
    swizzle_w_kernel<<<swgrd, swblk, 0, stream>>>(Wl2, nullptr, f_pap[l] + 1 * SEG, 128, 128);
    swizzle_w_kernel<<<swgrd, swblk, 0, stream>>>(Wr0, Wr2,    f_pap[l] + 2 * SEG, 128, 128);
    swizzle_w_kernel<<<swgrd, swblk, 0, stream>>>(Wl1, nullptr, f_aut[l] + 0 * SEG, 128, 128);
    swizzle_w_kernel<<<swgrd, swblk, 0, stream>>>(Wr1, nullptr, f_aut[l] + 1 * SEG, 128, 128);
    addvec_kernel<<<1, 256, 0, stream>>>(bl + (size_t)(l * 3 + 0) * HDIM,
                                         bl + (size_t)(l * 3 + 2) * HDIM,
                                         bsum[l], HDIM);
  }
  swizzle_w_kernel<<<swgrd, swblk, 0, stream>>>(lin_w, nullptr, f_fin, 128, 64);

  // ---- input projections (+ReLU) ----
  launch_gemm8(x_paper, nullptr, 256, nullptr, nullptr, 0, nullptr, nullptr, 0,
               f_projp, lin_p_b, yp0, NPAPER, FLAG_RELU, stream);
  launch_gemm8(x_author, nullptr, 128, nullptr, nullptr, 0, nullptr, nullptr, 0,
               f_proja, lin_a_b, ya0, NAUTHOR, FLAG_RELU, stream);

  float* yp = yp0; float* ypn = yp1;
  float* ya = ya0; float* yan = ya1;

  for (int l = 0; l < NLAYER; ++l) {
    hipMemsetAsync(s1, 0, NPH, stream);
    hipMemsetAsync(s2, 0, NPH, stream);
    hipMemsetAsync(s3, 0, NAH, stream);
    hipMemsetAsync(c1, 0, (size_t)NPAPER * sizeof(float), stream);
    hipMemsetAsync(c2, 0, (size_t)NPAPER * sizeof(float), stream);
    hipMemsetAsync(c3, 0, (size_t)NAUTHOR * sizeof(float), stream);

    scatter_add_kernel<<<sgrd, sblk, 0, stream>>>(ya, a2p_src, a2p_dst, s1, c1, NEDGE);
    scatter_add_kernel<<<sgrd, sblk, 0, stream>>>(yp, p2p_src, p2p_dst, s2, c2, NEDGE);
    scatter_add_kernel<<<sgrd, sblk, 0, stream>>>(yp, p2a_src, p2a_dst, s3, c3, NEDGE);

    // paper: new_p = [mean_a2p ; mean_p2p ; yp](M x 384) @ [Wl0;Wl2;Wr0+Wr2] + (bl0+bl2)
    launch_gemm8(s1, c1, 128, s2, c2, 128, yp, nullptr, 128,
                 f_pap[l], bsum[l], ypn, NPAPER, 0, stream);
    // author: new_a = [mean_p2a ; ya](M x 256) @ [Wl1;Wr1] + bl1
    launch_gemm8(s3, c3, 128, ya, nullptr, 128, nullptr, nullptr, 0,
                 f_aut[l], bl + (size_t)(l * 3 + 1) * HDIM, yan, NAUTHOR, 0, stream);

    float* t;
    t = yp; yp = ypn; ypn = t;
    t = ya; ya = yan; yan = t;
  }

  // ---- final classifier: [NP,128]@[128,64] + b ----
  {
    dim3 grid(((NPAPER + 31) / 32 + 7) / 8);
    wmma_gemm_kernel<4><<<grid, 256, 0, stream>>>(
        yp, nullptr, 128, nullptr, nullptr, 0, nullptr, nullptr, 0,
        f_fin, lin_b, (float*)d_out, NPAPER, 0);
  }
}
